// GATLayer_18657337934499
// MI455X (gfx1250) — compile-verified, matrix operand order checked by reading, not run on hardware
//
#include <hip/hip_runtime.h>
#include <hip/hip_bf16.h>

// Problem constants (from reference)
#define B_    4
#define N_    2048
#define IND   128
#define OUTD  128
#define H_    4
#define HD    32

constexpr float INV_TEMP = 5.0f;   // 1/0.2

typedef __attribute__((ext_vector_type(16))) __bf16 v16bf;
typedef __attribute__((ext_vector_type(2)))  __bf16 bf16x2;
typedef __attribute__((ext_vector_type(8)))  float  v8f;

typedef __attribute__((address_space(1))) int* gint_p;   // global int*
typedef __attribute__((address_space(3))) int* lint_p;   // LDS int*

__device__ __forceinline__ float lrelu(float x) { return x > 0.f ? x : 0.2f * x; }

// K index held by element pair (2v,2v+1) of the 16-bit WMMA A/B operand
// (ISA 7.12.2: lanes 0-15 -> K {0..7,16..23}; lanes 16-31 -> K {8..15,24..31})
__device__ __forceinline__ int wmma_kbase(int v, int lane) {
  return ((lane & 16) ? 8 : 0) + ((v < 4) ? (2 * v) : (16 + 2 * (v - 4)));
}

// ---------------------------------------------------------------- projection
// Wh[b][h][n][o] (f32) and WhT[b][h][o][m]=Wh[b][h][m][o] (bf16, B-operand layout)
__global__ void k_proj(const float* __restrict__ h, const float* __restrict__ W,
                       float* __restrict__ Wh, __bf16* __restrict__ WhT) {
  int idx = blockIdx.x * blockDim.x + threadIdx.x;   // B*H*N*HD = 1M
  int o  = idx & (HD - 1);
  int n  = (idx >> 5) & (N_ - 1);
  int bh = idx >> 16;
  int hh = bh & (H_ - 1);
  int b  = bh >> 2;
  const float* hp = h + (size_t)(b * N_ + n) * IND;
  const float* wp = W + (size_t)hh * IND * HD + o;
  float acc = 0.f;
#pragma unroll 8
  for (int i = 0; i < IND; ++i) acc = fmaf(hp[i], wp[(size_t)i * HD], acc);
  Wh[idx] = acc;
  WhT[((size_t)bh * HD + o) * N_ + n] = (__bf16)acc;
}

// ---------------------------------------------------------------- score vecs
__global__ void k_scores(const float* __restrict__ Wh, const float* __restrict__ a,
                         float* __restrict__ ssrc, float* __restrict__ sdst) {
  int idx = blockIdx.x * blockDim.x + threadIdx.x;   // B*H*N = 32768
  int bh = idx >> 11;
  int hh = bh & (H_ - 1);
  const float* w  = Wh + (size_t)idx * HD;
  const float* as = a + (size_t)hh * 2 * HD;
  const float* ad = as + HD;
  float s = 0.f, d = 0.f;
#pragma unroll
  for (int o = 0; o < HD; ++o) { s = fmaf(w[o], as[o], s); d = fmaf(w[o], ad[o], d); }
  ssrc[idx] = s;
  sdst[idx] = d;
}

// ---------------------------------------------------------------- max_m s_dst
__global__ void k_sdmax(const float* __restrict__ sdst, float* __restrict__ sdm) {
  __shared__ float red[256];
  const float* p = sdst + (size_t)blockIdx.x * N_;
  float m = -1e30f;
  for (int i = threadIdx.x; i < N_; i += 256) m = fmaxf(m, p[i]);
  red[threadIdx.x] = m;
  __syncthreads();
  for (int s = 128; s > 0; s >>= 1) {
    if ((int)threadIdx.x < s) red[threadIdx.x] = fmaxf(red[threadIdx.x], red[threadIdx.x + s]);
    __syncthreads();
  }
  if (threadIdx.x == 0) sdm[blockIdx.x] = red[0];
}

// ---------------------------------------------------------------- softmax denom
__global__ void k_rowsum(const float* __restrict__ ssrc, const float* __restrict__ sdst,
                         const float* __restrict__ sdm, float* __restrict__ rsum) {
  __shared__ float sd[N_];
  int bh = blockIdx.x >> 3;                        // 16 bh * 8 blocks
  int n  = ((blockIdx.x & 7) << 8) + threadIdx.x;
  const float* dp = sdst + (size_t)bh * N_;
  for (int i = threadIdx.x; i < N_; i += 256) sd[i] = dp[i];
  __syncthreads();
  float sv = ssrc[(size_t)bh * N_ + n];
  float rm = lrelu(sv + sdm[bh]) * INV_TEMP;       // closed-form row max
  float acc = 0.f;
  for (int m = 0; m < N_; ++m) acc += __expf(lrelu(sv + sd[m]) * INV_TEMP - rm);
  rsum[(size_t)bh * N_ + n] = acc;
}

// ---------------------------------------------------------------- bf16 convert
__global__ void k_cvt(const float* __restrict__ mw, __bf16* __restrict__ mwb) {
  int i = blockIdx.x * blockDim.x + threadIdx.x;
  if (i < OUTD * OUTD) mwb[i] = (__bf16)mw[i];
}

// ---------------------------------------------------------------- attention
// 1 block = 16 rows, 4 waves (1 head each). alpha chunk built in-register in the
// A-operand layout, normalized up-front, fed to v_wmma_f32_16x16x32_bf16 with
// WhT as B; head-mean accumulated in LDS and streamed out once.
__global__ void __launch_bounds__(128) k_attn(
    const float* __restrict__ ssrc, const float* __restrict__ sdst,
    const float* __restrict__ sdm, const float* __restrict__ rsum,
    const __bf16* __restrict__ WhT,
    float* __restrict__ alpha_out, __bf16* __restrict__ hc) {
  __shared__ float sd[H_][N_];        // 32 KB: s_dst per head
  __shared__ float acc[16][33];       // padded: avoid bank conflicts
  __shared__ float srow[H_][16], rmrow[H_][16], rsrow[H_][16];

  int b    = blockIdx.x >> 7;         // 128 row-tiles per batch
  int n0   = (blockIdx.x & 127) * 16;
  int lane = threadIdx.x & 31;
  int hh   = threadIdx.x >> 5;        // head = wave
  int bh   = b * H_ + hh;

  const float* dp = sdst + (size_t)bh * N_;
#if __has_builtin(__builtin_amdgcn_global_load_async_to_lds_b32)
  // Async DMA straight into LDS (ASYNCcnt path) - no VGPR round trip.
  for (int i = lane; i < N_; i += 32) {
    gint_p g = (gint_p)(uintptr_t)(dp + i);
    lint_p l = (lint_p)(uintptr_t)(&sd[hh][i]);
    __builtin_amdgcn_global_load_async_to_lds_b32(g, l, 0, 0);
  }
#if __has_builtin(__builtin_amdgcn_s_wait_asynccnt)
  __builtin_amdgcn_s_wait_asynccnt(0);
#else
  asm volatile("s_wait_asynccnt 0" ::: "memory");
#endif
#else
  for (int i = lane; i < N_; i += 32) sd[hh][i] = dp[i];
#endif
  if (lane < 16) {
    float sv = ssrc[(size_t)bh * N_ + n0 + lane];
    srow[hh][lane]  = sv;
    rmrow[hh][lane] = lrelu(sv + sdm[bh]) * INV_TEMP;
    rsrow[hh][lane] = 1.0f / rsum[(size_t)bh * N_ + n0 + lane];
  }
#pragma unroll
  for (int it = 0; it < 4; ++it) {
    int t = (int)threadIdx.x + it * 128;
    acc[t >> 5][t & 31] = 0.f;
  }
  __syncthreads();

  int   r   = lane & 15;              // A-operand row held by this lane
  float sv  = srow[hh][r];
  float rm  = rmrow[hh][r];
  float rsi = rsrow[hh][r];
  const __bf16* wb0 = WhT + ((size_t)bh * HD + r) * N_;        // B col r
  const __bf16* wb1 = WhT + ((size_t)bh * HD + 16 + r) * N_;   // B col 16+r

  v8f c0 = {}; v8f c1 = {};
  for (int mc = 0; mc < N_; mc += 32) {
    if (mc + 32 < N_) {               // cover B-operand latency of next chunk
      __builtin_prefetch(wb0 + mc + 32, 0, 1);   // global_prefetch_b8
      __builtin_prefetch(wb1 + mc + 32, 0, 1);
    }
    v16bf av, b0, b1;
#pragma unroll
    for (int v = 0; v < 8; ++v) {
      int k = wmma_kbase(v, lane);
      float p0 = __expf(lrelu(sv + sd[hh][mc + k])     * INV_TEMP - rm) * rsi;
      float p1 = __expf(lrelu(sv + sd[hh][mc + k + 1]) * INV_TEMP - rm) * rsi;
      av[2 * v]     = (__bf16)p0;
      av[2 * v + 1] = (__bf16)p1;
      atomicAdd(&acc[r][k], p0);       // ds_add_f32: sum over 4 heads
      atomicAdd(&acc[r][k + 1], p1);
      bf16x2 w0 = *(const bf16x2*)(wb0 + mc + k);   // b32 loads (pairs contiguous)
      bf16x2 w1 = *(const bf16x2*)(wb1 + mc + k);
      b0[2 * v] = w0[0]; b0[2 * v + 1] = w0[1];
      b1[2 * v] = w1[0]; b1[2 * v + 1] = w1[1];
    }
    c0 = __builtin_amdgcn_wmma_f32_16x16x32_bf16(false, av, false, b0, (short)0, c0, false, false);
    c1 = __builtin_amdgcn_wmma_f32_16x16x32_bf16(false, av, false, b1, (short)0, c1, false, false);
    __syncthreads();
#pragma unroll
    for (int it = 0; it < 4; ++it) {   // stream head-mean alpha (fixed trip: no exec-mask loop)
      int t  = (int)threadIdx.x + it * 128;
      int rr = t >> 5, kk = t & 31;
      alpha_out[(size_t)(b * N_ + n0 + rr) * N_ + mc + kk] = acc[rr][kk] * 0.25f;
      acc[rr][kk] = 0.f;
    }
    __syncthreads();
  }
  // D layout (ISA 7.12.2): VGPR rr, lane<16 -> (M=rr, N=lane); lane>=16 -> (M=rr+8)
#pragma unroll
  for (int rr = 0; rr < 8; ++rr) {
    int M = rr + ((lane & 16) ? 8 : 0);
    size_t row = (size_t)(b * N_ + n0 + M) * OUTD + hh * HD;
    hc[row + (lane & 15)]      = (__bf16)c0[rr];
    hc[row + 16 + (lane & 15)] = (__bf16)c1[rr];
  }
}

// ---------------------------------------------------------------- merge GEMM
__global__ void __launch_bounds__(128) k_merge(
    const __bf16* __restrict__ hc, const __bf16* __restrict__ mwb,
    const float* __restrict__ bias, float* __restrict__ out) {
  int b    = blockIdx.x >> 7;
  int n0   = (blockIdx.x & 127) * 16;
  int lane = threadIdx.x & 31;
  int j0   = (threadIdx.x >> 5) * 32;   // each wave owns 32 output cols
  int r    = lane & 15;
  const __bf16* arow  = hc  + (size_t)(b * N_ + n0 + r) * OUTD;
  const __bf16* brow0 = mwb + (size_t)(j0 + r) * OUTD;       // B[k][j]=merge_w[j][k]
  const __bf16* brow1 = mwb + (size_t)(j0 + 16 + r) * OUTD;
  v8f c0 = {}; v8f c1 = {};
  for (int kc = 0; kc < OUTD; kc += 32) {
    v16bf av, b0, b1;
#pragma unroll
    for (int v = 0; v < 8; ++v) {
      int k = wmma_kbase(v, lane);
      bf16x2 a2 = *(const bf16x2*)(arow  + kc + k);
      bf16x2 w0 = *(const bf16x2*)(brow0 + kc + k);
      bf16x2 w1 = *(const bf16x2*)(brow1 + kc + k);
      av[2 * v] = a2[0]; av[2 * v + 1] = a2[1];
      b0[2 * v] = w0[0]; b0[2 * v + 1] = w0[1];
      b1[2 * v] = w1[0]; b1[2 * v + 1] = w1[1];
    }
    c0 = __builtin_amdgcn_wmma_f32_16x16x32_bf16(false, av, false, b0, (short)0, c0, false, false);
    c1 = __builtin_amdgcn_wmma_f32_16x16x32_bf16(false, av, false, b1, (short)0, c1, false, false);
  }
#pragma unroll
  for (int rr = 0; rr < 8; ++rr) {
    int M = rr + ((lane & 16) ? 8 : 0);
    int j = j0 + (lane & 15);
    size_t row = (size_t)(b * N_ + n0 + M) * OUTD;
    out[row + j]      = c0[rr] + bias[j];
    out[row + j + 16] = c1[rr] + bias[j + 16];
  }
}

extern "C" void kernel_launch(void* const* d_in, const int* in_sizes, int n_in,
                              void* d_out, int out_size, void* d_ws, size_t ws_size,
                              hipStream_t stream) {
  const float* h  = (const float*)d_in[0];
  const float* W  = (const float*)d_in[1];
  const float* a  = (const float*)d_in[2];
  const float* mw = (const float*)d_in[3];
  const float* mb = (const float*)d_in[4];

  char* ws = (char*)d_ws;                       // ~8.8 MB used
  float*  Wh   = (float*)(ws + 0);              // 4 MB  f32 Wh
  __bf16* WhT  = (__bf16*)(ws + 4194304);       // 2 MB  bf16 Wh^T [b][h][o][m]
  float*  ssrc = (float*)(ws + 6291456);        // 128 KB
  float*  sdst = (float*)(ws + 6422528);        // 128 KB
  float*  sdm  = (float*)(ws + 6553600);        // 256 B
  float*  rsum = (float*)(ws + 6553856);        // 128 KB
  __bf16* hc   = (__bf16*)(ws + 6684928);       // 2 MB  bf16 concat heads
  __bf16* mwb  = (__bf16*)(ws + 8782080);       // 32 KB bf16 merge_w

  float* hp = (float*)d_out;                    // h_prime: B*N*OUTD f32
  float* al = hp + (size_t)B_ * N_ * OUTD;      // alpha mean: B*N*N f32

  k_proj  <<<4096, 256, 0, stream>>>(h, W, Wh, WhT);
  k_scores<<<128,  256, 0, stream>>>(Wh, a, ssrc, sdst);
  k_sdmax <<<16,   256, 0, stream>>>(sdst, sdm);
  k_rowsum<<<128,  256, 0, stream>>>(ssrc, sdst, sdm, rsum);
  k_cvt   <<<64,   256, 0, stream>>>(mw, mwb);
  k_attn  <<<512,  128, 0, stream>>>(ssrc, sdst, sdm, rsum, WhT, al, hc);
  k_merge <<<512,  128, 0, stream>>>(hc, mwb, mb, hp);
}
